// AdaptiveNoisingModule_44289702756645
// MI455X (gfx1250) — matrix operand order repaired and checked by reading, not compiled, not run on hardware
//
#include <hip/hip_runtime.h>
#include <hip/hip_bf16.h>
#include <math.h>
#include <stdint.h>

typedef __attribute__((ext_vector_type(16))) __bf16 v16bf;
typedef __attribute__((ext_vector_type(8)))  __bf16 v8bf;
typedef __attribute__((ext_vector_type(8)))  float  v8f;

#define N_ROWS   6272
#define DIM      1024
#define M_ROWS   16384
#define HW       784
#define NSEG     8
#define SEGCOLS  2048
#define KNB      9

// workspace byte offsets (all 256B aligned)
#define XT_OFF  ((size_t)0)          // x transposed fp32 [N,D]          25,690,112 B
#define XB_OFF  ((size_t)25690112)   // x bf16 [N,D]                     12,845,056 B
#define MB_OFF  ((size_t)38535168)   // memory bank bf16 [M,D]           33,554,432 B
#define XN_OFF  ((size_t)72089600)   // ||x||^2 [N]
#define MN_OFF  ((size_t)72114688)   // ||m||^2 [M]
#define DS_OFF  ((size_t)72180224)   // mean top-9 distance [N]
#define PD_OFF  ((size_t)72205312)   // partial top-9 dist [N][NSEG][9]
#define PI_OFF  ((size_t)74011648)   // partial top-9 idx  [N][NSEG][9]
#define IDX_OFF ((size_t)75817984)   // final top-9 idx [N][9]
#define ST_OFF  ((size_t)76043776)   // {full_mean, full_std}

#define OUT_IMAP 6422528
#define OUT_NMAP 6428800

__device__ __forceinline__ float block_reduce_sum(float v, float* red) {
  const int t = threadIdx.x;
  red[t] = v;
  __syncthreads();
#pragma unroll
  for (int off = 128; off > 0; off >>= 1) {
    if (t < off) red[t] += red[t + off];
    __syncthreads();
  }
  float r = red[0];
  __syncthreads();
  return r;
}

__device__ __forceinline__ void topk_insert(float (&td)[KNB], int (&ti)[KNB],
                                            float v, int c) {
  if (v < td[KNB - 1]) {
    td[KNB - 1] = v; ti[KNB - 1] = c;
#pragma unroll
    for (int j = KNB - 1; j > 0; --j) {
      if (td[j] < td[j - 1]) {
        float tf = td[j]; td[j] = td[j - 1]; td[j - 1] = tf;
        int tc = ti[j];   ti[j] = ti[j - 1]; ti[j - 1] = tc;
      }
    }
  }
}

// ---------------- LDS-tiled NCHW -> [N,D] transpose (+bf16 copy) ------------
// grid (25 p-tiles, 32 d-tiles, 8 batches), block 256 (tx=lane in p/d, ty=row)
__global__ void __launch_bounds__(256) transpose_kernel(
    const float* __restrict__ feat, float* __restrict__ xT,
    __bf16* __restrict__ xb) {
  __shared__ float tile[32][33];
  const int tx = threadIdx.x & 31, ty = threadIdx.x >> 5;
  const int b = blockIdx.z, dt = blockIdx.y, pt = blockIdx.x;
#pragma unroll
  for (int i = 0; i < 4; ++i) {
    const int dl = ty + i * 8;
    const int p = pt * 32 + tx;
    float v = 0.f;
    if (p < HW)
      v = feat[(size_t)b * DIM * HW + (size_t)(dt * 32 + dl) * HW + p];
    tile[dl][tx] = v;  // coalesced read along p
  }
  __syncthreads();
#pragma unroll
  for (int i = 0; i < 4; ++i) {
    const int pl = ty + i * 8;
    const int p = pt * 32 + pl;
    if (p < HW) {
      const int n = b * HW + p;
      const int d = dt * 32 + tx;
      const float v = tile[tx][pl];          // transposed, no bank conflicts
      xT[(size_t)n * DIM + d] = v;           // coalesced write along d
      xb[(size_t)n * DIM + d] = (__bf16)v;
    }
  }
}

__global__ void __launch_bounds__(256) xnorm_kernel(
    const float* __restrict__ xT, float* __restrict__ xnorm) {
  __shared__ float red[256];
  const int n = blockIdx.x, t = threadIdx.x;
  float ss = 0.f;
#pragma unroll
  for (int j = 0; j < 4; ++j) {
    float f = xT[(size_t)n * DIM + t + j * 256];
    ss += f * f;
  }
  float tot = block_reduce_sum(ss, red);
  if (t == 0) xnorm[n] = tot;
}

__global__ void __launch_bounds__(256) prep_m_kernel(
    const float* __restrict__ memb, __bf16* __restrict__ mbB,
    float* __restrict__ mnorm) {
  __shared__ float red[256];
  const int m = blockIdx.x, t = threadIdx.x;
  float ss = 0.f;
#pragma unroll
  for (int j = 0; j < 4; ++j) {
    int d = t + j * 256;
    float f = memb[(size_t)m * DIM + d];
    mbB[(size_t)m * DIM + d] = (__bf16)f;
    ss += f * f;
  }
  float tot = block_reduce_sum(ss, red);
  if (t == 0) mnorm[m] = tot;
}

// ---------------- WMMA GEMM + async B staging + streaming top-9 -------------
// grid (49, 8): blockIdx.x = 128-row block, blockIdx.y = 2048-col segment.
// B chunks (64 cols x full K, 128KB bf16) double-buffered in LDS via
// GLOBAL_LOAD_ASYNC_TO_LDS_B128 (ASYNCcnt); 8 waves consume each tile.
__global__ void __launch_bounds__(256) gemm_topk_kernel(
    const __bf16* __restrict__ xb, const __bf16* __restrict__ mbB,
    const float* __restrict__ xnorm, const float* __restrict__ mnorm,
    float* __restrict__ pd, int* __restrict__ pi) {
  __shared__ __align__(32) __bf16 bbuf[2][64][DIM];  // 2 x 128 KB
  __shared__ float distbuf[8][16][64];               // 32 KB
  const int tid  = threadIdx.x;
  const int wave = tid >> 5, lane = tid & 31;
  const int lr   = lane & 15, lh = lane >> 4;
  const int rowBase = blockIdx.x * 128 + wave * 16;
  const int seg     = blockIdx.y;
  const int mStart  = seg * SEGCOLS;

  float xn[8];
#pragma unroll
  for (int j = 0; j < 8; ++j) xn[j] = xnorm[rowBase + lh * 8 + j];

  const __bf16* aRow = xb + (size_t)(rowBase + lr) * DIM;

  float td[KNB]; int ti[KNB];
#pragma unroll
  for (int k = 0; k < KNB; ++k) { td[k] = 3.0e38f; ti[k] = 0; }

  const v8f zf = {0.f, 0.f, 0.f, 0.f, 0.f, 0.f, 0.f, 0.f};

  // async copy of one 64-col x 1024-K bf16 chunk (contiguous 128KB) into LDS
  auto stage = [&](int ci, int bi) {
    const __bf16* gsrc = mbB + (size_t)(mStart + ci * 64) * DIM;
    const unsigned lbase = (unsigned)(uintptr_t)(&bbuf[bi][0][0]);
#pragma unroll 4
    for (int u = 0; u < 32; ++u) {
      const int unit = tid + u * 256;  // 8192 x 16B units
      const uint64_t ga = (uint64_t)(uintptr_t)(gsrc + (size_t)unit * 8);
      const unsigned la = lbase + (unsigned)unit * 16;
      asm volatile("global_load_async_to_lds_b128 %0, %1, off"
                   :: "v"(la), "v"(ga) : "memory");
    }
  };

  stage(0, 0);
  for (int ch = 0; ch < SEGCOLS / 64; ++ch) {
    const int colBase = mStart + ch * 64;
    if (ch + 1 < SEGCOLS / 64) {
      stage(ch + 1, (ch + 1) & 1);
      // async loads complete in order: oldest 32 (= previous chunk) retired
      asm volatile("s_wait_asynccnt 0x20" ::: "memory");
    } else {
      asm volatile("s_wait_asynccnt 0x0" ::: "memory");
    }
    __syncthreads();  // all threads' stage(ch) writes visible

    const __bf16* bt = &bbuf[ch & 1][0][0];
    v8f acc[4];
#pragma unroll
    for (int t = 0; t < 4; ++t) acc[t] = zf;

#pragma unroll 2
    for (int k0 = 0; k0 < DIM; k0 += 32) {
      // A fragment: lane = M row, K split across lane halves (contiguous 16)
      v16bf A = *reinterpret_cast<const v16bf*>(aRow + k0 + lh * 16);
      // B fragments from LDS: lane = N col, K split across lane halves
      v16bf B0 = *reinterpret_cast<const v16bf*>(bt + (size_t)( 0 + lr) * DIM + k0 + lh * 16);
      v16bf B1 = *reinterpret_cast<const v16bf*>(bt + (size_t)(16 + lr) * DIM + k0 + lh * 16);
      v16bf B2 = *reinterpret_cast<const v16bf*>(bt + (size_t)(32 + lr) * DIM + k0 + lh * 16);
      v16bf B3 = *reinterpret_cast<const v16bf*>(bt + (size_t)(48 + lr) * DIM + k0 + lh * 16);
      acc[0] = __builtin_amdgcn_wmma_f32_16x16x32_bf16(false, A, false, B0,
                                                       (short)0, acc[0], false, false);
      acc[1] = __builtin_amdgcn_wmma_f32_16x16x32_bf16(false, A, false, B1,
                                                       (short)0, acc[1], false, false);
      acc[2] = __builtin_amdgcn_wmma_f32_16x16x32_bf16(false, A, false, B2,
                                                       (short)0, acc[2], false, false);
      acc[3] = __builtin_amdgcn_wmma_f32_16x16x32_bf16(false, A, false, B3,
                                                       (short)0, acc[3], false, false);
    }

    // d = sqrt(clip(||x||^2 + ||m||^2 - 2 x.m)), staged to LDS
#pragma unroll
    for (int t = 0; t < 4; ++t) {
      const int m = colBase + t * 16 + lr;
      const float mn = mnorm[m];
#pragma unroll
      for (int j = 0; j < 8; ++j) {
        float d2 = xn[j] + mn - 2.0f * acc[t][j];
        float dd = sqrtf(fmaxf(d2, 1e-12f));
        distbuf[wave][lh * 8 + j][t * 16 + lr] = dd;
      }
    }
    // per-row streaming top-9: lane scans row lr, half lh of 64 cols
#pragma unroll 4
    for (int s = 0; s < 32; ++s) {
      float v = distbuf[wave][lr][lh * 32 + s];
      topk_insert(td, ti, v, colBase + lh * 32 + s);
    }
    __syncthreads();  // everyone done with bbuf[ch&1] before it is re-staged
  }

  // merge the two half-lane lists per row through LDS
  if (lane >= 16) {
#pragma unroll
    for (int k = 0; k < KNB; ++k) {
      distbuf[wave][lane - 16][k] = td[k];
      distbuf[wave][lane - 16][KNB + k] = __int_as_float(ti[k]);
    }
  }
  __syncthreads();
  if (lane < 16) {
#pragma unroll
    for (int k = 0; k < KNB; ++k)
      topk_insert(td, ti, distbuf[wave][lr][k],
                  __float_as_int(distbuf[wave][lr][KNB + k]));
    const int row = rowBase + lr;
    const size_t base = ((size_t)row * NSEG + seg) * KNB;
#pragma unroll
    for (int k = 0; k < KNB; ++k) { pd[base + k] = td[k]; pi[base + k] = ti[k]; }
  }
}

// ---------------- merge segment lists, mean top-9 distance ------------------
__global__ void __launch_bounds__(256) merge_topk_kernel(
    const float* __restrict__ pd, const int* __restrict__ pi,
    float* __restrict__ ds, int* __restrict__ idx) {
  const int row = blockIdx.x * 256 + threadIdx.x;
  if (row >= N_ROWS) return;
  float td[KNB]; int ti[KNB];
#pragma unroll
  for (int k = 0; k < KNB; ++k) { td[k] = 3.0e38f; ti[k] = 0; }
  for (int s = 0; s < NSEG; ++s) {
    const size_t base = ((size_t)row * NSEG + s) * KNB;
#pragma unroll
    for (int k = 0; k < KNB; ++k) topk_insert(td, ti, pd[base + k], pi[base + k]);
  }
  float sum = 0.f;
#pragma unroll
  for (int k = 0; k < KNB; ++k) { sum += td[k]; idx[row * KNB + k] = ti[k]; }
  ds[row] = sum * (1.0f / (float)KNB);
}

// ---------------- global distance-signal stats ------------------------------
__global__ void __launch_bounds__(256) ds_stats_kernel(
    const float* __restrict__ ds, float* __restrict__ st) {
  __shared__ float red[256];
  __shared__ float meanSh;
  const int t = threadIdx.x;
  float s = 0.f;
  for (int i = t; i < N_ROWS; i += 256) s += ds[i];
  float tot = block_reduce_sum(s, red);
  if (t == 0) meanSh = tot / (float)N_ROWS;
  __syncthreads();
  const float m = meanSh;
  float s2 = 0.f;
  for (int i = t; i < N_ROWS; i += 256) { float d = ds[i] - m; s2 += d * d; }
  float tot2 = block_reduce_sum(s2, red);
  if (t == 0) {
    st[0] = m;
    st[1] = sqrtf(tot2 * (float)DIM / ((float)N_ROWS * (float)DIM - 1.0f));
  }
}

// ---------------- influence, normalization, noise, outputs ------------------
__global__ void __launch_bounds__(256) finalize_kernel(
    const float* __restrict__ xT, const float* __restrict__ memb,
    const float* __restrict__ iw, const float* __restrict__ dw,
    const float* __restrict__ eps, const float* __restrict__ ds,
    const int* __restrict__ idx, const float* __restrict__ st,
    float* __restrict__ out) {
  __shared__ float red[256];
  __shared__ int sidx[KNB];
  const int n = blockIdx.x;
  const int t = threadIdx.x;
  if (t < KNB) sidx[t] = idx[n * KNB + t];
  __syncthreads();
  const float dn  = (ds[n] - st[0]) / (st[1] + 1e-8f);
  const float dwv = dw[0];

  float infl[4], xs[4];
  float s1 = 0.f, s2 = 0.f;
#pragma unroll
  for (int j = 0; j < 4; ++j) {
    const int d = t + j * 256;
    const float xv = xT[(size_t)n * DIM + d];
    xs[j] = xv;
    float a = 0.f;
#pragma unroll
    for (int k = 0; k < KNB; ++k)
      a += fabsf(xv - memb[(size_t)sidx[k] * DIM + d]);
    const float v = (a * (1.0f / (float)KNB)) * iw[d];
    infl[j] = v; s1 += v; s2 += v * v;
  }
  const float S1 = block_reduce_sum(s1, red);
  const float S2 = block_reduce_sum(s2, red);
  const float mean = S1 / (float)DIM;
  const float var  = (S2 - (float)DIM * mean * mean) / ((float)DIM - 1.0f);
  const float stdv = sqrtf(fmaxf(var, 0.f));

  const int b = n / HW, p = n % HW;
  float nsum = 0.f;
#pragma unroll
  for (int j = 0; j < 4; ++j) {
    const int d = t + j * 256;
    const float inorm = (infl[j] - mean) / (stdv + 1e-8f);
    const float comb  = inorm + dwv * dn;
    const float sig   = 1.0f / (1.0f + expf(-comb));
    const float ns    = 0.01f + 0.49f * sig;
    nsum += ns;
    out[(size_t)b * DIM * HW + (size_t)d * HW + p] =
        xs[j] + eps[(size_t)n * DIM + d] * ns;
  }
  const float NS = block_reduce_sum(nsum, red);
  if (t == 0) {
    out[OUT_IMAP + n] = mean;            // influence.mean(axis=-1)
    out[OUT_NMAP + n] = NS / (float)DIM; // noise_std.mean(axis=-1)
  }
}

extern "C" void kernel_launch(void* const* d_in, const int* in_sizes, int n_in,
                              void* d_out, int out_size, void* d_ws, size_t ws_size,
                              hipStream_t stream) {
  const float* feat = (const float*)d_in[0];
  const float* memb = (const float*)d_in[1];
  const float* iw   = (const float*)d_in[2];
  const float* dw   = (const float*)d_in[3];
  const float* eps  = (const float*)d_in[4];
  char* ws = (char*)d_ws;

  float*  xT  = (float*)(ws + XT_OFF);
  __bf16* xb  = (__bf16*)(ws + XB_OFF);
  __bf16* mbB = (__bf16*)(ws + MB_OFF);
  float*  xn  = (float*)(ws + XN_OFF);
  float*  mn  = (float*)(ws + MN_OFF);
  float*  ds  = (float*)(ws + DS_OFF);
  float*  pd  = (float*)(ws + PD_OFF);
  int*    pi  = (int*)(ws + PI_OFF);
  int*    idx = (int*)(ws + IDX_OFF);
  float*  st  = (float*)(ws + ST_OFF);
  float*  out = (float*)d_out;

  transpose_kernel<<<dim3(25, 32, 8), 256, 0, stream>>>(feat, xT, xb);
  xnorm_kernel<<<N_ROWS, 256, 0, stream>>>(xT, xn);
  prep_m_kernel<<<M_ROWS, 256, 0, stream>>>(memb, mbB, mn);
  dim3 g(N_ROWS / 128, NSEG);
  gemm_topk_kernel<<<g, 256, 0, stream>>>(xb, mbB, xn, mn, pd, pi);
  merge_topk_kernel<<<(N_ROWS + 255) / 256, 256, 0, stream>>>(pd, pi, ds, idx);
  ds_stats_kernel<<<1, 256, 0, stream>>>(ds, st);
  finalize_kernel<<<N_ROWS, 256, 0, stream>>>(xT, memb, iw, dw, eps, ds, idx, st, out);
}
